// GeometryAwareCrossAttentionBlock_63728724738252
// MI455X (gfx1250) — compile-verified, matrix operand order checked by reading, not run on hardware
//
#include <hip/hip_runtime.h>
#include <math.h>

// ---------------- types / helpers ----------------
typedef __attribute__((ext_vector_type(16))) __bf16 v16bf;
typedef __attribute__((ext_vector_type(4)))  __bf16 v4bf;
typedef __attribute__((ext_vector_type(8)))  float  v8f;

struct alignas(16) U4 { unsigned int x, y, z, w; };
struct U8 { U4 lo, hi; };
struct alignas(16) F4 { float x, y, z, w; };

#define DEV static __device__ __forceinline__

DEV unsigned short f2bf(float f) {                 // native f32->bf16 convert
  __bf16 b = (__bf16)f;
  return __builtin_bit_cast(unsigned short, b);
}
DEV unsigned long long pack4(float a, float b, float c, float d) {
  v4bf v = { (__bf16)a, (__bf16)b, (__bf16)c, (__bf16)d };  // -> v_cvt_pk_bf16_f32 x2
  return __builtin_bit_cast(unsigned long long, v);
}
// A/B fragment for v_wmma_f32_16x16x32_bf16 from a row-major bf16 LDS row.
// lane: idx = lane&15 (m for A, n for B); kb = (lane>>4)*8.
// halves 0..7 = row[kb..kb+7], halves 8..15 = row[kb+16..kb+23].
DEV v16bf load_frag(const unsigned short* rowp, int kb) {
  U8 t;
  t.lo = *(const U4*)(rowp + kb);
  t.hi = *(const U4*)(rowp + kb + 16);
  return __builtin_bit_cast(v16bf, t);
}
DEV v8f wmma_bf16(v16bf a, v16bf b, v8f c) {
  return __builtin_amdgcn_wmma_f32_16x16x32_bf16(false, a, false, b, (short)0, c, false, false);
}
DEV v8f vzero() { v8f z = {0.f,0.f,0.f,0.f,0.f,0.f,0.f,0.f}; return z; }
DEV float gelu_tanh(float x) {
  float x3 = x * x * x;
  return 0.5f * x * (1.0f + tanhf(0.7978845608f * (x + 0.044715f * x3)));
}

// ---------------- weight transpose+convert: W(K,N) f32 -> Wt(N,K) bf16 ----------------
__global__ void prep_w(const float* __restrict__ W, unsigned short* __restrict__ Wt,
                       int K, int N) {
  int i = blockIdx.x * 256 + threadIdx.x;
  if (i >= K * N) return;
  int n = i / K, k = i - n * K;
  Wt[i] = f2bf(W[(size_t)k * N + n]);
}

// ---------------- LayerNorm over C=384, one row per block (128 thr) ----------------
__global__ __launch_bounds__(128) void ln_kernel(const float* __restrict__ x,
                                                 const float* __restrict__ g,
                                                 const float* __restrict__ bb,
                                                 float* __restrict__ out) {
  int row = blockIdx.x, t = threadIdx.x;
  const float* xr = x + (size_t)row * 384;
  float v0 = xr[t], v1 = xr[t + 128], v2 = xr[t + 256];
  float s = v0 + v1 + v2, s2 = v0*v0 + v1*v1 + v2*v2;
  #pragma unroll
  for (int m = 1; m < 32; m <<= 1) { s += __shfl_xor(s, m, 32); s2 += __shfl_xor(s2, m, 32); }
  __shared__ float ss[4], sq[4];
  int wid = t >> 5;
  if ((t & 31) == 0) { ss[wid] = s; sq[wid] = s2; }
  __syncthreads();
  s = ss[0] + ss[1] + ss[2] + ss[3];
  s2 = sq[0] + sq[1] + sq[2] + sq[3];
  float mean = s * (1.0f / 384.0f);
  float var  = s2 * (1.0f / 384.0f) - mean * mean;
  float rstd = rsqrtf(var + 1e-5f);
  float* orow = out + (size_t)row * 384;
  orow[t]       = (v0 - mean) * rstd * g[t]       + bb[t];
  orow[t + 128] = (v1 - mean) * rstd * g[t + 128] + bb[t + 128];
  orow[t + 256] = (v2 - mean) * rstd * g[t + 256] + bb[t + 256];
}

// ---------------- generic bf16-WMMA GEMM ----------------
// out[row,col] = act( sum_k A(row,k)*W(k,col) + bias[col] ) + resid[row,col]
// A: k<splitK from A1 else A2 (dual-A fuses the concat). Wt is bf16 (N,K) row-major.
__global__ __launch_bounds__(128) void gemm_kernel(
    const float* __restrict__ A1, const float* __restrict__ A2, int splitK, int lda,
    const unsigned short* __restrict__ Wt, int ldw,
    const float* __restrict__ bias, const float* __restrict__ resid, int ldr,
    float* __restrict__ out, int ldo, int Ktot, int act) {
  __shared__ __align__(16) unsigned short As[64 * 32];
  __shared__ __align__(16) unsigned short Bs[64 * 32];
  int t = threadIdx.x, w = t >> 5, lane = t & 31;
  int rowBase = blockIdx.x * 64, colBase = blockIdx.y * 64;
  v8f acc[4];
  #pragma unroll
  for (int j = 0; j < 4; ++j) acc[j] = vzero();

  for (int k0 = 0; k0 < Ktot; k0 += 32) {
    const float* Ap; int kk;
    if (k0 < splitK) { Ap = A1; kk = k0; } else { Ap = A2; kk = k0 - splitK; }
    F4 av[4]; U4 bv[2];
    #pragma unroll
    for (int i = 0; i < 4; ++i) {                         // issue A loads (batched)
      int idx4 = t + i * 128;
      int r = idx4 >> 3, c4 = idx4 & 7;
      av[i] = *(const F4*)(Ap + (size_t)(rowBase + r) * lda + kk + c4 * 4);
    }
    #pragma unroll
    for (int i = 0; i < 2; ++i) {                         // issue B loads (batched)
      int idx8 = t + i * 128;
      int n = idx8 >> 2, g = idx8 & 3;
      bv[i] = *(const U4*)(Wt + (size_t)(colBase + n) * ldw + k0 + g * 8);
    }
    if (k0 + 32 < Ktot) {                                 // gfx1250 global_prefetch
      __builtin_prefetch(Wt + (size_t)(colBase + (t & 63)) * ldw + k0 + 32, 0, 1);
      __builtin_prefetch(Ap + (size_t)(rowBase + (t & 63)) * lda + kk + 32, 0, 1);
    }
    #pragma unroll
    for (int i = 0; i < 4; ++i) {                         // convert + store A
      int idx4 = t + i * 128;
      int r = idx4 >> 3, c4 = idx4 & 7;
      *(unsigned long long*)(&As[r * 32 + c4 * 4]) =
          pack4(av[i].x, av[i].y, av[i].z, av[i].w);
    }
    #pragma unroll
    for (int i = 0; i < 2; ++i) {                         // store B
      int idx8 = t + i * 128;
      int n = idx8 >> 2, g = idx8 & 3;
      *(U4*)(&Bs[n * 32 + g * 8]) = bv[i];
    }
    __syncthreads();
    int m = lane & 15, kb = (lane >> 4) * 8;
    v16bf af = load_frag(&As[(w * 16 + m) * 32], kb);
    #pragma unroll
    for (int j = 0; j < 4; ++j) {
      v16bf bf = load_frag(&Bs[(j * 16 + m) * 32], kb);
      acc[j] = wmma_bf16(af, bf, acc[j]);
    }
    __syncthreads();
  }
  int nn = lane & 15, hh = lane >> 4;
  #pragma unroll
  for (int j = 0; j < 4; ++j) {
    int col = colBase + j * 16 + nn;
    float bv2 = bias ? bias[col] : 0.0f;
    #pragma unroll
    for (int r = 0; r < 8; ++r) {
      int row = rowBase + w * 16 + r + 8 * hh;
      float v = acc[j][r] + bv2;
      if (act == 1) v = gelu_tanh(v);
      if (resid) v += resid[(size_t)row * ldr + col];
      out[(size_t)row * ldo + col] = v;
    }
  }
}

// ---------------- KNN top-8 (per batch), branchless register insertion ----------------
__global__ __launch_bounds__(256) void knn_kernel(const float* __restrict__ qc,
                                                  const float* __restrict__ kc,
                                                  int* __restrict__ idxout) {
  __shared__ float skx[256], sky[256], skz[256], sn2[256];
  int b = blockIdx.y;
  int q = blockIdx.x * 256 + threadIdx.x;
  const float* qp = qc + ((size_t)b * 2048 + q) * 3;
  float qx = qp[0], qy = qp[1], qz = qp[2];
  float bd[8]; int bi[8];
  #pragma unroll
  for (int i = 0; i < 8; ++i) { bd[i] = 3.4e38f; bi[i] = 0; }
  for (int kt = 0; kt < 2048; kt += 256) {
    __syncthreads();
    {
      const float* kp = kc + ((size_t)b * 2048 + kt + threadIdx.x) * 3;
      float x = kp[0], y = kp[1], z = kp[2];
      skx[threadIdx.x] = x; sky[threadIdx.x] = y; skz[threadIdx.x] = z;
      sn2[threadIdx.x] = x * x + y * y + z * z;
    }
    __syncthreads();
    for (int j = 0; j < 256; ++j) {
      float d = sn2[j] - 2.0f * (qx * skx[j] + qy * sky[j] + qz * skz[j]);
      if (d < bd[7]) {
        int ni = kt + j;
        #pragma unroll
        for (int i = 7; i >= 1; --i) {
          bool sh = d < bd[i - 1];
          bool here = !sh && (d < bd[i]);
          float nv = sh ? bd[i - 1] : (here ? d : bd[i]);
          int   nx = sh ? bi[i - 1] : (here ? ni : bi[i]);
          bd[i] = nv; bi[i] = nx;
        }
        if (d < bd[0]) { bd[0] = d; bi[0] = ni; }
      }
    }
  }
  #pragma unroll
  for (int i = 0; i < 8; ++i)
    idxout[((size_t)b * 2048 + q) * 8 + i] = bi[i];
}

// ---------------- flash attention per (b,h), 64-query tiles ----------------
__global__ __launch_bounds__(128) void attn_kernel(
    const float* __restrict__ qp, const float* __restrict__ kp,
    const float* __restrict__ vp, int ld, float* __restrict__ outp) {
  __shared__ __align__(16) unsigned short Qs[64 * 64];
  __shared__ __align__(16) unsigned short Ks[64 * 64];
  __shared__ __align__(16) unsigned short Vts[64 * 64];   // [d][key]
  __shared__ __align__(16) unsigned short Ps[4][16 * 64]; // per-wave P tile
  int t = threadIdx.x, w = t >> 5, lane = t & 31;
  int h = blockIdx.y % 6, b = blockIdx.y / 6;
  int qBase = blockIdx.x * 64;
  size_t rowOff = (size_t)b * 2048;
  {                                                       // stage Q (batched loads)
    F4 qv[8];
    #pragma unroll
    for (int i = 0; i < 8; ++i) {
      int idx4 = t + i * 128;
      int r = idx4 >> 4, c4 = idx4 & 15;
      qv[i] = *(const F4*)(qp + (rowOff + qBase + r) * ld + h * 64 + c4 * 4);
    }
    #pragma unroll
    for (int i = 0; i < 8; ++i) {
      int idx4 = t + i * 128;
      int r = idx4 >> 4, c4 = idx4 & 15;
      *(unsigned long long*)(&Qs[r * 64 + c4 * 4]) =
          pack4(qv[i].x, qv[i].y, qv[i].z, qv[i].w);
    }
  }
  __syncthreads();
  int m = lane & 15, kb = (lane >> 4) * 8, nn = lane & 15, hh = lane >> 4;
  v16bf qf[2];
  qf[0] = load_frag(&Qs[(w * 16 + m) * 64], kb);
  qf[1] = load_frag(&Qs[(w * 16 + m) * 64 + 32], kb);
  v8f oacc[4];
  #pragma unroll
  for (int j = 0; j < 4; ++j) oacc[j] = vzero();
  float m8[8], l8[8];
  #pragma unroll
  for (int r = 0; r < 8; ++r) { m8[r] = -3.4e38f; l8[r] = 0.0f; }

  for (int kt = 0; kt < 2048; kt += 64) {
    __syncthreads();
    #pragma unroll
    for (int half = 0; half < 2; ++half) {                // stage K (batches of 4)
      F4 kv[4];
      #pragma unroll
      for (int i = 0; i < 4; ++i) {
        int idx4 = t + (half * 4 + i) * 128;
        int r = idx4 >> 4, c4 = idx4 & 15;
        kv[i] = *(const F4*)(kp + (rowOff + kt + r) * ld + h * 64 + c4 * 4);
      }
      #pragma unroll
      for (int i = 0; i < 4; ++i) {
        int idx4 = t + (half * 4 + i) * 128;
        int r = idx4 >> 4, c4 = idx4 & 15;
        *(unsigned long long*)(&Ks[r * 64 + c4 * 4]) =
            pack4(kv[i].x, kv[i].y, kv[i].z, kv[i].w);
      }
    }
    #pragma unroll
    for (int half = 0; half < 2; ++half) {                // stage V^T (batches of 4)
      F4 vv[4];
      #pragma unroll
      for (int i = 0; i < 4; ++i) {
        int idx4 = t + (half * 4 + i) * 128;
        int r = idx4 >> 4, c4 = idx4 & 15;
        vv[i] = *(const F4*)(vp + (rowOff + kt + r) * ld + h * 64 + c4 * 4);
      }
      #pragma unroll
      for (int i = 0; i < 4; ++i) {
        int idx4 = t + (half * 4 + i) * 128;
        int r = idx4 >> 4, c4 = idx4 & 15;
        Vts[(c4 * 4 + 0) * 64 + r] = f2bf(vv[i].x);
        Vts[(c4 * 4 + 1) * 64 + r] = f2bf(vv[i].y);
        Vts[(c4 * 4 + 2) * 64 + r] = f2bf(vv[i].z);
        Vts[(c4 * 4 + 3) * 64 + r] = f2bf(vv[i].w);
      }
    }
    __syncthreads();
    v8f sacc[4];
    #pragma unroll
    for (int j = 0; j < 4; ++j) sacc[j] = vzero();
    #pragma unroll
    for (int ks = 0; ks < 2; ++ks) {                      // S = Q K^T
      #pragma unroll
      for (int j = 0; j < 4; ++j) {
        v16bf bf = load_frag(&Ks[(j * 16 + nn) * 64 + ks * 32], kb);
        sacc[j] = wmma_bf16(qf[ks], bf, sacc[j]);
      }
    }
    #pragma unroll
    for (int r = 0; r < 8; ++r) {                         // online softmax (rows in 16-lane halves)
      float rm = -3.4e38f;
      #pragma unroll
      for (int j = 0; j < 4; ++j) { sacc[j][r] *= 0.125f; rm = fmaxf(rm, sacc[j][r]); }
      rm = fmaxf(rm, __shfl_xor(rm, 1, 32));
      rm = fmaxf(rm, __shfl_xor(rm, 2, 32));
      rm = fmaxf(rm, __shfl_xor(rm, 4, 32));
      rm = fmaxf(rm, __shfl_xor(rm, 8, 32));
      float mn = fmaxf(m8[r], rm);
      float al = __expf(m8[r] - mn);
      m8[r] = mn;
      float rs = 0.0f;
      #pragma unroll
      for (int j = 0; j < 4; ++j) {
        float p = __expf(sacc[j][r] - mn);
        rs += p;
        Ps[w][(r + 8 * hh) * 64 + j * 16 + nn] = f2bf(p);
      }
      rs += __shfl_xor(rs, 1, 32);
      rs += __shfl_xor(rs, 2, 32);
      rs += __shfl_xor(rs, 4, 32);
      rs += __shfl_xor(rs, 8, 32);
      l8[r] = l8[r] * al + rs;
      #pragma unroll
      for (int j = 0; j < 4; ++j) oacc[j][r] *= al;
    }
    #pragma unroll
    for (int ks = 0; ks < 2; ++ks) {                      // O += P V
      v16bf pf = load_frag(&Ps[w][m * 64 + ks * 32], kb);
      #pragma unroll
      for (int j = 0; j < 4; ++j) {
        v16bf vf = load_frag(&Vts[(j * 16 + nn) * 64 + ks * 32], kb);
        oacc[j] = wmma_bf16(pf, vf, oacc[j]);
      }
    }
  }
  #pragma unroll
  for (int j = 0; j < 4; ++j)
    #pragma unroll
    for (int r = 0; r < 8; ++r) {
      int row = qBase + w * 16 + r + 8 * hh;
      outp[(rowOff + row) * 384 + h * 64 + j * 16 + nn] = oacc[j][r] / l8[r];
    }
}

// ---------------- gathered edge GEMM + leakyReLU + max over 8 neighbors ----------------
// tile = 8 queries x 8 neighbors (64 rows); geom[q,col] = max_kk lrelu(edge + cproj)
__global__ __launch_bounds__(128) void edge_kernel(
    const unsigned short* __restrict__ Wt, int ldw,
    const float* __restrict__ kf, const float* __restrict__ cf,
    const int* __restrict__ knn, const float* __restrict__ cproj,
    float* __restrict__ geom) {
  __shared__ __align__(16) unsigned short As[64 * 32];
  __shared__ __align__(16) unsigned short Bs[64 * 32];
  __shared__ int sidx[64];
  int t = threadIdx.x, w = t >> 5, lane = t & 31;
  int qBase = blockIdx.x * 8;          // global query row (0..4095)
  int colBase = blockIdx.y * 64;
  int b = qBase >> 11;
  if (t < 64) {
    int q = qBase + (t >> 3), kk = t & 7;
    sidx[t] = (b << 11) + knn[(size_t)q * 8 + kk];
  }
  __syncthreads();
  v8f acc[4];
  #pragma unroll
  for (int j = 0; j < 4; ++j) acc[j] = vzero();

  for (int k0 = 0; k0 < 384; k0 += 32) {
    F4 av[4], cv[4]; U4 bv[2];
    #pragma unroll
    for (int i = 0; i < 4; ++i) {                         // issue gather + center loads
      int idx4 = t + i * 128;
      int r = idx4 >> 3, c4 = idx4 & 7;
      int q = qBase + (r >> 3);
      av[i] = *(const F4*)(kf + (size_t)sidx[r] * 384 + k0 + c4 * 4);
      cv[i] = *(const F4*)(cf + (size_t)q * 384 + k0 + c4 * 4);
    }
    #pragma unroll
    for (int i = 0; i < 2; ++i) {
      int idx8 = t + i * 128;
      int n = idx8 >> 2, g = idx8 & 3;
      bv[i] = *(const U4*)(Wt + (size_t)(colBase + n) * ldw + k0 + g * 8);
    }
    #pragma unroll
    for (int i = 0; i < 4; ++i) {                         // subtract, convert, store
      int idx4 = t + i * 128;
      int r = idx4 >> 3, c4 = idx4 & 7;
      *(unsigned long long*)(&As[r * 32 + c4 * 4]) =
          pack4(av[i].x - cv[i].x, av[i].y - cv[i].y,
                av[i].z - cv[i].z, av[i].w - cv[i].w);
    }
    #pragma unroll
    for (int i = 0; i < 2; ++i) {
      int idx8 = t + i * 128;
      int n = idx8 >> 2, g = idx8 & 3;
      *(U4*)(&Bs[n * 32 + g * 8]) = bv[i];
    }
    __syncthreads();
    int m = lane & 15, kb = (lane >> 4) * 8;
    v16bf af = load_frag(&As[(w * 16 + m) * 32], kb);
    #pragma unroll
    for (int j = 0; j < 4; ++j) {
      v16bf bf = load_frag(&Bs[(j * 16 + m) * 32], kb);
      acc[j] = wmma_bf16(af, bf, acc[j]);
    }
    __syncthreads();
  }
  int nn = lane & 15, hh = lane >> 4;
  int q = qBase + w * 2 + hh;                              // in-lane rows = 8 neighbors of q
  #pragma unroll
  for (int j = 0; j < 4; ++j) {
    int col = colBase + j * 16 + nn;
    float cp = cproj[(size_t)q * 384 + col];
    float mx = -3.4e38f;
    #pragma unroll
    for (int r = 0; r < 8; ++r) {
      float e = acc[j][r] + cp;
      e = e > 0.0f ? e : 0.2f * e;
      mx = fmaxf(mx, e);
    }
    geom[(size_t)q * 384 + col] = mx;
  }
}

// ---------------- host orchestration ----------------
extern "C" void kernel_launch(void* const* d_in, const int* in_sizes, int n_in,
                              void* d_out, int out_size, void* d_ws, size_t ws_size,
                              hipStream_t stream) {
  (void)in_sizes; (void)n_in; (void)out_size; (void)ws_size;
  const float* qc    = (const float*)d_in[0];
  const float* xf    = (const float*)d_in[1];
  const float* kcrd  = (const float*)d_in[2];
  const float* kfeat = (const float*)d_in[3];
  const float* in_w  = (const float*)d_in[4];
  const float* in_b  = (const float*)d_in[5];
  const float* W_qkv = (const float*)d_in[6];
  const float* W_mo  = (const float*)d_in[7];
  const float* b_mo  = (const float*)d_in[8];
  const float* W_sg  = (const float*)d_in[9];
  const float* b_sg  = (const float*)d_in[10];
  const float* W_sm  = (const float*)d_in[11];
  const float* b_sm  = (const float*)d_in[12];
  const float* cnq_w = (const float*)d_in[13];
  const float* cnq_b = (const float*)d_in[14];
  const float* cnk_w = (const float*)d_in[15];
  const float* cnk_b = (const float*)d_in[16];
  const float* W_cq  = (const float*)d_in[17];
  const float* W_ck  = (const float*)d_in[18];
  const float* W_cv  = (const float*)d_in[19];
  const float* W_xo  = (const float*)d_in[20];
  const float* b_xo  = (const float*)d_in[21];
  const float* W_cg  = (const float*)d_in[22];
  const float* b_cg  = (const float*)d_in[23];
  const float* W_cm  = (const float*)d_in[24];
  const float* b_cm  = (const float*)d_in[25];
  const float* fn_w  = (const float*)d_in[26];
  const float* fn_b  = (const float*)d_in[27];
  const float* W_f1  = (const float*)d_in[28];
  const float* b_f1  = (const float*)d_in[29];
  const float* W_f2  = (const float*)d_in[30];
  const float* b_f2  = (const float*)d_in[31];
  float* outp = (float*)d_out;

  char* ws = (char*)d_ws;
  size_t off = 0;
  auto alloc = [&](size_t bytes) -> char* {
    char* p = ws + off;
    off = (off + bytes + 255) & ~(size_t)255;
    return p;
  };
  const int BIG = 1 << 30;
  // bf16 transposed weights
  unsigned short* wt_qkv = (unsigned short*)alloc((size_t)384 * 1152 * 2);
  unsigned short* wt_mo  = (unsigned short*)alloc((size_t)384 * 384 * 2);
  unsigned short* wt_sg  = (unsigned short*)alloc((size_t)768 * 384 * 2);
  unsigned short* wt_sm  = (unsigned short*)alloc((size_t)768 * 384 * 2);
  unsigned short* wt_cq  = (unsigned short*)alloc((size_t)384 * 384 * 2);
  unsigned short* wt_ck  = (unsigned short*)alloc((size_t)384 * 384 * 2);
  unsigned short* wt_cv  = (unsigned short*)alloc((size_t)384 * 384 * 2);
  unsigned short* wt_xo  = (unsigned short*)alloc((size_t)384 * 384 * 2);
  unsigned short* wt_cg  = (unsigned short*)alloc((size_t)768 * 384 * 2);
  unsigned short* wt_cm  = (unsigned short*)alloc((size_t)768 * 384 * 2);
  unsigned short* wt_f1  = (unsigned short*)alloc((size_t)384 * 768 * 2);
  unsigned short* wt_f2  = (unsigned short*)alloc((size_t)768 * 384 * 2);
  // f32 activation buffers (M = B*N = 4096 rows)
  float* nf    = (float*)alloc((size_t)4096 * 384 * 4);   // normed (reused as nq, ffn h)
  float* qkvb  = (float*)alloc((size_t)4096 * 1152 * 4);  // qkv; reused cq/ck/cv; reused hid
  float* attnO = (float*)alloc((size_t)4096 * 384 * 4);
  float* attnP = (float*)alloc((size_t)4096 * 384 * 4);
  float* cproj = (float*)alloc((size_t)4096 * 384 * 4);
  float* geom  = (float*)alloc((size_t)4096 * 384 * 4);
  float* nk    = (float*)alloc((size_t)4096 * 384 * 4);
  int*   idxb  = (int*)alloc((size_t)4096 * 8 * 4);
  float* cqb = qkvb;
  float* ckb = qkvb + (size_t)4096 * 384;
  float* cvb = qkvb + (size_t)4096 * 768;
  float* hid = qkvb;

  auto prep = [&](const float* W, unsigned short* Wt, int K, int N) {
    int tot = K * N;
    prep_w<<<dim3((tot + 255) / 256), dim3(256), 0, stream>>>(W, Wt, K, N);
  };
  prep(W_qkv, wt_qkv, 384, 1152);
  prep(W_mo,  wt_mo,  384, 384);
  prep(W_sg,  wt_sg,  768, 384);
  prep(W_sm,  wt_sm,  768, 384);
  prep(W_cq,  wt_cq,  384, 384);
  prep(W_ck,  wt_ck,  384, 384);
  prep(W_cv,  wt_cv,  384, 384);
  prep(W_xo,  wt_xo,  384, 384);
  prep(W_cg,  wt_cg,  768, 384);
  prep(W_cm,  wt_cm,  768, 384);
  prep(W_f1,  wt_f1,  384, 768);
  prep(W_f2,  wt_f2,  768, 384);

  dim3 blk(128);
  // ---- phase 1: self ----
  ln_kernel<<<dim3(4096), blk, 0, stream>>>(xf, in_w, in_b, nf);
  gemm_kernel<<<dim3(64, 18), blk, 0, stream>>>(nf, nf, BIG, 384, wt_qkv, 384,
      nullptr, nullptr, 0, qkvb, 1152, 384, 0);
  knn_kernel<<<dim3(8, 2), dim3(256), 0, stream>>>(qc, qc, idxb);
  gemm_kernel<<<dim3(64, 6), blk, 0, stream>>>(nf, nf, BIG, 384, wt_sg + 384, 768,
      b_sg, nullptr, 0, cproj, 384, 384, 0);
  attn_kernel<<<dim3(32, 12), blk, 0, stream>>>(qkvb, qkvb + 384, qkvb + 768, 1152, attnO);
  gemm_kernel<<<dim3(64, 6), blk, 0, stream>>>(attnO, attnO, BIG, 384, wt_mo, 384,
      b_mo, nullptr, 0, attnP, 384, 384, 0);
  edge_kernel<<<dim3(512, 6), blk, 0, stream>>>(wt_sg, 768, nf, nf, idxb, cproj, geom);
  gemm_kernel<<<dim3(64, 6), blk, 0, stream>>>(attnP, geom, 384, 384, wt_sm, 768,
      b_sm, xf, 384, outp, 384, 768, 0);
  // ---- phase 2: cross ----
  ln_kernel<<<dim3(4096), blk, 0, stream>>>(outp, cnq_w, cnq_b, nf);
  ln_kernel<<<dim3(4096), blk, 0, stream>>>(kfeat, cnk_w, cnk_b, nk);
  gemm_kernel<<<dim3(64, 6), blk, 0, stream>>>(nf, nf, BIG, 384, wt_cq, 384,
      nullptr, nullptr, 0, cqb, 384, 384, 0);
  gemm_kernel<<<dim3(64, 6), blk, 0, stream>>>(nk, nk, BIG, 384, wt_ck, 384,
      nullptr, nullptr, 0, ckb, 384, 384, 0);
  gemm_kernel<<<dim3(64, 6), blk, 0, stream>>>(nk, nk, BIG, 384, wt_cv, 384,
      nullptr, nullptr, 0, cvb, 384, 384, 0);
  knn_kernel<<<dim3(8, 2), dim3(256), 0, stream>>>(qc, kcrd, idxb);
  gemm_kernel<<<dim3(64, 6), blk, 0, stream>>>(nf, nf, BIG, 384, wt_cg + 384, 768,
      b_cg, nullptr, 0, cproj, 384, 384, 0);
  attn_kernel<<<dim3(32, 12), blk, 0, stream>>>(cqb, ckb, cvb, 384, attnO);
  gemm_kernel<<<dim3(64, 6), blk, 0, stream>>>(attnO, attnO, BIG, 384, wt_xo, 384,
      b_xo, nullptr, 0, attnP, 384, 384, 0);
  edge_kernel<<<dim3(512, 6), blk, 0, stream>>>(wt_cg, 768, nk, nf, idxb, cproj, geom);
  gemm_kernel<<<dim3(64, 6), blk, 0, stream>>>(attnP, geom, 384, 384, wt_cm, 768,
      b_cm, outp, 384, outp, 384, 768, 0);
  // ---- phase 3: FFN ----
  ln_kernel<<<dim3(4096), blk, 0, stream>>>(outp, fn_w, fn_b, nf);
  gemm_kernel<<<dim3(64, 12), blk, 0, stream>>>(nf, nf, BIG, 384, wt_f1, 384,
      b_f1, nullptr, 0, hid, 768, 384, 1);
  gemm_kernel<<<dim3(64, 6), blk, 0, stream>>>(hid, hid, BIG, 768, wt_f2, 768,
      b_f2, outp, 384, outp, 384, 768, 0);
}
